// OCCNet_52029233823911
// MI455X (gfx1250) — matrix operand order repaired and checked by reading, not compiled, not run on hardware
//
#include <hip/hip_runtime.h>
#include <hip/hip_bf16.h>
#include <math.h>

typedef __attribute__((ext_vector_type(16))) _Float16 v16h;
typedef __attribute__((ext_vector_type(8)))  _Float16 v8h;
typedef __attribute__((ext_vector_type(8)))  float    v8f;

#define HTABLE   8192
#define NLODS    16
#define BLOCK    256
#define LDH      72   // padded LDS row stride in halfs: 144B = 9*16B -> 16B-aligned rows

// load 16 halfs as two aligned 16B chunks -> one v16h fragment
__device__ __forceinline__ v16h ld2x8(const _Float16* p0, const _Float16* p1) {
    v8h x = *reinterpret_cast<const v8h*>(p0);
    v8h y = *reinterpret_cast<const v8h*>(p1);
    return __builtin_shufflevector(x, y, 0,1,2,3,4,5,6,7,8,9,10,11,12,13,14,15);
}

// One 16-row x 64-col tile of  relu(X[16x64] @ W[64x64] + b), in-place in `io`.
// A layout (16b 16x32): lane<16 -> K-halves 0..7 (+16 in VGPR4-7); lane>=16 -> 8..15 (+16).
// B layout (16b 32x16): lane = column, same K order as A (8-K runs at 8h and 16+8h).
__device__ __forceinline__ void tile_layer(_Float16 (*io)[LDH],
                                           const _Float16 (*wT)[LDH],
                                           const float* bias,
                                           int tb, int lane) {
    const int lh = lane & 15;
    const int hi = (lane >> 4) & 1;
    const int m  = tb + lh;
    const int ka = hi ? 8 : 0;
    v16h a0 = ld2x8(&io[m][ka],      &io[m][16 + ka]);   // K = 0..31
    v16h a1 = ld2x8(&io[m][32 + ka], &io[m][48 + ka]);   // K = 32..63
    const int ks = hi ? 8 : 0;
    v8f acc[4];
#pragma unroll
    for (int nt = 0; nt < 4; ++nt) {
        const int n = nt * 16 + lh;
        const float bv = bias[n];
#pragma unroll
        for (int v = 0; v < 8; ++v) acc[nt][v] = bv;   // bias broadcast per column
        v16h bf0 = ld2x8(&wT[n][ks], &wT[n][16 + ks]);
        acc[nt] = __builtin_amdgcn_wmma_f32_16x16x32_f16(false, a0, false, bf0,
                                                         (short)0, acc[nt], false, false);
        v16h bf1 = ld2x8(&wT[n][32 + ks], &wT[n][48 + ks]);
        acc[nt] = __builtin_amdgcn_wmma_f32_16x16x32_f16(false, a1, false, bf1,
                                                         (short)0, acc[nt], false, false);
    }
    __builtin_amdgcn_wave_barrier();    // all A reads precede in-place writes
    const int mb = tb + (hi ? 8 : 0);   // C/D layout: VGPR v -> row mb+v, lane -> column
#pragma unroll
    for (int nt = 0; nt < 4; ++nt) {
        const int n = nt * 16 + lh;
#pragma unroll
        for (int v = 0; v < 8; ++v) {
            float x = acc[nt][v];
            x = x > 0.0f ? x : 0.0f;    // ReLU
            io[mb + v][n] = (_Float16)x;
        }
    }
    __builtin_amdgcn_wave_barrier();
}

__global__ __launch_bounds__(BLOCK)
void OCCNet_52029233823911_kernel(const float* __restrict__ pts,
                                  const float* __restrict__ tables,
                                  const float* __restrict__ w0, const float* __restrict__ b0,
                                  const float* __restrict__ w1, const float* __restrict__ b1,
                                  const float* __restrict__ w2, const float* __restrict__ b2,
                                  float* __restrict__ out) {
    __shared__ __align__(16) _Float16 sF[BLOCK][LDH];   // feats -> h1 -> h2 (reused)
    __shared__ __align__(16) _Float16 sW0[64][LDH];     // w0^T as [n][k] f16
    __shared__ __align__(16) _Float16 sW1[64][LDH];     // w1^T as [n][k] f16
    __shared__ float sB0[64], sB1[64], sW2[64];
    __shared__ float sB2;

    const int tid  = threadIdx.x;
    const int lane = tid & 31;
    const int wid  = tid >> 5;
    const int gid  = blockIdx.x * BLOCK + tid;

    // ---- stage MLP weights into LDS (f16, transposed) ----
    for (int i = tid; i < 64 * 64; i += BLOCK) {
        const int k = i >> 6, n = i & 63;   // source is row-major [k][n]
        sW0[n][k] = (_Float16)w0[i];
        sW1[n][k] = (_Float16)w1[i];
    }
    if (tid < 64) { sB0[tid] = b0[tid]; sB1[tid] = b1[tid]; sW2[tid] = w2[tid]; }
    if (tid == 0) sB2 = b2[0];

    // ---- Phase A: hash-grid trilinear features (16 LODs x 4 feats = 64) ----
    const float px = pts[gid * 3 + 0];
    const float py = pts[gid * 3 + 1];
    const float pz = pts[gid * 3 + 2];
    const float4* __restrict__ tab = reinterpret_cast<const float4*>(tables);

    for (int l = 0; l < NLODS; ++l) {
        const int res   = 1 + (16 << l);         // b = 2 exactly for this config
        const float rm1 = (float)(res - 1);
        float fx = px * rm1, fy = py * rm1, fz = pz * rm1;
        int ix = (int)floorf(fx), iy = (int)floorf(fy), iz = (int)floorf(fz);
        const int cmax = res - 2;
        ix = ix < 0 ? 0 : (ix > cmax ? cmax : ix);
        iy = iy < 0 ? 0 : (iy > cmax ? cmax : iy);
        iz = iz < 0 ? 0 : (iz > cmax ? cmax : iz);
        fx -= (float)ix; fy -= (float)iy; fz -= (float)iz;
        const float4* __restrict__ tl = tab + l * HTABLE;
        float a0 = 0.f, a1 = 0.f, a2 = 0.f, a3 = 0.f;
#pragma unroll
        for (int cc = 0; cc < 8; ++cc) {
            const int ci = (cc >> 2) & 1, cj = (cc >> 1) & 1, ck = cc & 1;
            const unsigned cx = (unsigned)(ix + ci);
            const unsigned cy = (unsigned)(iy + cj);
            const unsigned cz = (unsigned)(iz + ck);
            unsigned idx;
            if (l == 0) idx = (cx * 17u + cy) * 17u + cz;              // dense: 17^3 <= 8192
            else idx = (cx ^ (cy * 2654435761u) ^ (cz * 805459861u)) & (HTABLE - 1);
            const float w = (ci ? fx : 1.f - fx) * (cj ? fy : 1.f - fy) * (ck ? fz : 1.f - fz);
            const float4 v = tl[idx];                                   // hot in L2 (2MB tables)
            a0 += w * v.x; a1 += w * v.y; a2 += w * v.z; a3 += w * v.w;
        }
        sF[tid][4 * l + 0] = (_Float16)a0;
        sF[tid][4 * l + 1] = (_Float16)a1;
        sF[tid][4 * l + 2] = (_Float16)a2;
        sF[tid][4 * l + 3] = (_Float16)a3;
    }
    __syncthreads();

    // ---- Phase B: MLP layers 0 and 1 on WMMA; each wave owns 32 rows (2 tiles) ----
    for (int t = 0; t < 2; ++t) tile_layer(sF, sW0, sB0, wid * 32 + t * 16, lane);
    __syncthreads();
    for (int t = 0; t < 2; ++t) tile_layer(sF, sW1, sB1, wid * 32 + t * 16, lane);
    __syncthreads();

    // ---- Layer 2: 64 -> 1 dot + sigmoid, one thread per point ----
    float acc = sB2;
#pragma unroll
    for (int k = 0; k < 64; ++k) acc += (float)sF[tid][k] * sW2[k];
    out[gid] = 1.0f / (1.0f + __expf(-acc));
}

extern "C" void kernel_launch(void* const* d_in, const int* in_sizes, int n_in,
                              void* d_out, int out_size, void* d_ws, size_t ws_size,
                              hipStream_t stream) {
    (void)n_in; (void)d_ws; (void)ws_size; (void)out_size;
    const float* pts    = (const float*)d_in[0];
    const float* tables = (const float*)d_in[1];
    const float* w0     = (const float*)d_in[2];
    const float* b0     = (const float*)d_in[3];
    const float* w1     = (const float*)d_in[4];
    const float* b1     = (const float*)d_in[5];
    const float* w2     = (const float*)d_in[6];
    const float* b2     = (const float*)d_in[7];
    float* out = (float*)d_out;

    const int N = in_sizes[0] / 3;          // 1,048,576 points
    const int blocks = N / BLOCK;           // divisible: 4096 blocks
    OCCNet_52029233823911_kernel<<<blocks, BLOCK, 0, stream>>>(
        pts, tables, w0, b0, w1, b1, w2, b2, out);
}
